// BaichuanAttention_32847909880361
// MI455X (gfx1250) — compile-verified, matrix operand order checked by reading, not run on hardware
//
#include <hip/hip_runtime.h>
#include <hip/hip_bf16.h>
#include <math.h>

// ---------------------------------------------------------------------------
// Baichuan attention block for MI455X (gfx1250, wave32, WMMA).
// Pipeline: f32->bf16 convert -> QKV GEMM (wmma bf16, async-to-LDS double
// buffered) -> RoPE/split/V-transpose -> flash attention (wmma bf16, online
// softmax) -> O-proj GEMM.  Workspace usage: ~384 MB of d_ws.
// ---------------------------------------------------------------------------

typedef __attribute__((ext_vector_type(16))) __bf16 v16bf;
typedef __attribute__((ext_vector_type(8)))  __bf16 v8bf;
typedef __attribute__((ext_vector_type(8)))  float  v8f;

constexpr int BATCH  = 2;
constexpr int SEQLEN = 2048;
constexpr int HID    = 4096;
constexpr int NH     = 32;
constexpr int HD     = 128;
constexpr int MTOT   = BATCH * SEQLEN;   // 4096 rows
constexpr int N_QKV  = 3 * HID;          // 12288

__device__ __forceinline__ v8f v8f_zero() {
  v8f z;
#pragma unroll
  for (int i = 0; i < 8; ++i) z[i] = 0.0f;
  return z;
}

// 16 contiguous bf16 as two 16B chunks (works for 16B-aligned LDS/global ptrs)
__device__ __forceinline__ v16bf load16(const __bf16* p) {
  union { v16bf v; v8bf h[2]; } u;
  u.h[0] = *(const v8bf*)(p);
  u.h[1] = *(const v8bf*)(p + 8);
  return u.v;
}

// Wave-relative LDS byte offset: flat-LDS aperture keeps it in addr[31:0].
__device__ __forceinline__ unsigned lds_u32(const void* p) {
  return (unsigned)(unsigned long long)p;
}

// Async DMA: 16B per lane, global -> LDS, tracked by ASYNCcnt.
__device__ __forceinline__ void async_load_b128(unsigned lds_off, const void* gaddr) {
  asm volatile("global_load_async_to_lds_b128 %0, %1, off"
               :: "v"(lds_off), "v"(gaddr) : "memory");
}

__device__ __forceinline__ void wait_asynccnt0() {
  asm volatile("s_wait_asynccnt 0" ::: "memory");
}

// ---------------------------------------------------------------------------
// fp32 -> bf16 convert (grid stride)
// ---------------------------------------------------------------------------
__global__ void cvt_f32_to_bf16(const float* __restrict__ in,
                                __bf16* __restrict__ out, size_t n) {
  size_t i = (size_t)blockIdx.x * blockDim.x + threadIdx.x;
  const size_t stride = (size_t)gridDim.x * blockDim.x;
  for (; i < n; i += stride) out[i] = (__bf16)in[i];
}

// ---------------------------------------------------------------------------
// GEMM: C[M,N] = A[M,K] @ Bw[N,K]^T, bf16 in, f32 accumulate, OutT out.
// Block 256 threads (8 waves, 2x4), tile 128x128x32.
// Tiles are DMA'd to LDS with global_load_async_to_lds_b128 (ASYNCcnt),
// double buffered so the DMA of tile k+1 overlaps WMMA on tile k.
// Both operands are K-contiguous => every WMMA fragment load is a contiguous
// 32B LDS read per lane (A: lane=M row, B: lane=N row of Bw = column of B).
// ---------------------------------------------------------------------------
template <typename OutT>
__global__ void __launch_bounds__(256)
gemm_bf16_nt(const __bf16* __restrict__ A, const __bf16* __restrict__ Bw,
             OutT* __restrict__ C, int M, int N, int Kd) {
  __shared__ __bf16 As[2][128 * 40];   // +8 pad/row to stagger LDS banks
  __shared__ __bf16 Bs[2][128 * 40];
  const int t    = threadIdx.x;
  const int wave = t >> 5, lane = t & 31;
  const int wm = wave & 1, wn = wave >> 1;      // wave grid 2(M) x 4(N)
  const int lr = lane & 15, lh = lane >> 4;     // tile row/col, K half
  const size_t mBase = (size_t)blockIdx.y * 128;
  const size_t nBase = (size_t)blockIdx.x * 128;
  const int ldRow = t >> 1;                     // 0..127
  const int ldSeg = (t & 1) * 16;               // 0 or 16 bf16
  const __bf16* Ag = A  + (mBase + ldRow) * (size_t)Kd + ldSeg;
  const __bf16* Bg = Bw + (nBase + ldRow) * (size_t)Kd + ldSeg;
  const unsigned aOff[2] = { lds_u32(&As[0][ldRow * 40 + ldSeg]),
                             lds_u32(&As[1][ldRow * 40 + ldSeg]) };
  const unsigned bOff[2] = { lds_u32(&Bs[0][ldRow * 40 + ldSeg]),
                             lds_u32(&Bs[1][ldRow * 40 + ldSeg]) };

  auto issue_tile = [&](int buf, int kt) {
    async_load_b128(aOff[buf],      Ag + kt);
    async_load_b128(aOff[buf] + 16, Ag + kt + 8);   // +8 bf16 = +16 bytes
    async_load_b128(bOff[buf],      Bg + kt);
    async_load_b128(bOff[buf] + 16, Bg + kt + 8);
  };

  v8f acc[4][2];
#pragma unroll
  for (int i = 0; i < 4; ++i)
#pragma unroll
    for (int j = 0; j < 2; ++j) acc[i][j] = v8f_zero();

  // prologue: fill buffer 0
  issue_tile(0, 0);
  wait_asynccnt0();
  __syncthreads();

  int cur = 0;
  for (int kt = 0; kt < Kd; kt += 32) {
    if (kt + 32 < Kd) {
      issue_tile(cur ^ 1, kt + 32);               // DMA next tile
      __builtin_prefetch(Ag + kt + 64, 0, 1);     // global_prefetch_b8 (L2)
      __builtin_prefetch(Bg + kt + 64, 0, 1);
    }

    const __bf16* Asc = As[cur];
    const __bf16* Bsc = Bs[cur];
    v16bf af[4], bfm[2];
#pragma unroll
    for (int mt = 0; mt < 4; ++mt)
      af[mt] = load16(Asc + (wm * 64 + mt * 16 + lr) * 40 + lh * 16);
#pragma unroll
    for (int nt = 0; nt < 2; ++nt)
      bfm[nt] = load16(Bsc + (wn * 32 + nt * 16 + lr) * 40 + lh * 16);
#pragma unroll
    for (int mt = 0; mt < 4; ++mt)
#pragma unroll
      for (int nt = 0; nt < 2; ++nt)
        acc[mt][nt] = __builtin_amdgcn_wmma_f32_16x16x32_bf16(
            false, af[mt], false, bfm[nt], (short)0, acc[mt][nt], false, false);

    wait_asynccnt0();      // next tile fully landed in LDS
    __syncthreads();       // all waves done reading 'cur'; next data visible
    cur ^= 1;
  }

  // C layout: VGPR r, lanes 0-15 -> M=r, lanes 16-31 -> M=8+r; N=lane%16
#pragma unroll
  for (int mt = 0; mt < 4; ++mt)
#pragma unroll
    for (int nt = 0; nt < 2; ++nt)
#pragma unroll
      for (int r = 0; r < 8; ++r) {
        const size_t row = mBase + wm * 64 + mt * 16 + lh * 8 + r;
        const size_t col = nBase + wn * 32 + nt * 16 + lr;
        C[row * (size_t)N + col] = (OutT)acc[mt][nt][r];
      }
}

// ---------------------------------------------------------------------------
// RoPE (neox half-split) on q/k + head split; V written transposed [b,h,d,s]
// so PV WMMA B-fragments become contiguous loads. One block per (b,h,s).
// ---------------------------------------------------------------------------
__global__ void __launch_bounds__(128)
rope_split_kernel(const __bf16* __restrict__ qkv, const int* __restrict__ positions,
                  __bf16* __restrict__ Qo, __bf16* __restrict__ Ko,
                  __bf16* __restrict__ Vto) {
  const int idx = blockIdx.x;
  const int s = idx % SEQLEN;
  const int h = (idx / SEQLEN) % NH;
  const int b = idx / (SEQLEN * NH);
  const int t = threadIdx.x;                    // 0..127

  const size_t rowbase = ((size_t)(b * SEQLEN + s)) * N_QKV + (size_t)h * HD;
  const __bf16* qrow = qkv + rowbase;
  const __bf16* krow = qrow + HID;
  const __bf16* vrow = qrow + 2 * HID;
  const size_t obase = (((size_t)b * NH + h) * SEQLEN + s) * HD;

  if (t < 64) {
    const float pos = (float)positions[b * SEQLEN + s];
    // inv_freq[t] = 10000^(-t/64) = exp(-t * ln(10000)/64)
    const float f = pos * __expf(-(float)t * (9.210340371976184f / 64.0f));
    float sn, cs;
    sincosf(f, &sn, &cs);
    const float q1 = (float)qrow[t], q2 = (float)qrow[t + 64];
    Qo[obase + t]      = (__bf16)(q1 * cs - q2 * sn);
    Qo[obase + t + 64] = (__bf16)(q2 * cs + q1 * sn);
    const float k1 = (float)krow[t], k2 = (float)krow[t + 64];
    Ko[obase + t]      = (__bf16)(k1 * cs - k2 * sn);
    Ko[obase + t + 64] = (__bf16)(k2 * cs + k1 * sn);
  }
  Vto[(((size_t)b * NH + h) * HD + t) * SEQLEN + s] = vrow[t];
}

// ---------------------------------------------------------------------------
// Flash attention: block = 4 waves, 64 q rows (16/wave), KV blocks of 64.
// GEMM1: S = Q K^T (wmma bf16, f32 acc), online softmax in C-layout,
// P staged bf16 through per-wave LDS (C-layout -> A-layout), GEMM2: O += P V.
// ---------------------------------------------------------------------------
__global__ void __launch_bounds__(128)
flash_attn_kernel(const __bf16* __restrict__ Q, const __bf16* __restrict__ K,
                  const __bf16* __restrict__ Vt, __bf16* __restrict__ attn_out) {
  const int qb   = blockIdx.x;                  // 64-row q block
  const int h    = blockIdx.y;
  const int b    = blockIdx.z;
  const int wave = threadIdx.x >> 5;            // 0..3
  const int lane = threadIdx.x & 31;
  const int lr   = lane & 15;
  const int lh   = lane >> 4;

  const size_t bh = (size_t)b * NH + h;
  const __bf16* Qp = Q  + bh * SEQLEN * HD;
  const __bf16* Kp = K  + bh * SEQLEN * HD;
  const __bf16* Vp = Vt + bh * HD * SEQLEN;

  __shared__ __bf16 Plds[4][16][72];            // per-wave P tile, +8 pad

  // Preload Q A-fragments (rows fixed for whole kernel): 4 chunks of hd=128
  const int qrowA = qb * 64 + wave * 16 + lr;
  v16bf aq[4];
#pragma unroll
  for (int ks = 0; ks < 4; ++ks)
    aq[ks] = load16(Qp + (size_t)qrowA * HD + ks * 32 + lh * 16);

  v8f o[8];
#pragma unroll
  for (int i = 0; i < 8; ++i) o[i] = v8f_zero();
  float m_r[8], l_r[8];
#pragma unroll
  for (int r = 0; r < 8; ++r) { m_r[r] = -1e30f; l_r[r] = 0.0f; }

  const float scale = 0.08838834764831845f;     // 1/sqrt(128)
  const int qrow0 = qb * 64 + wave * 16 + lh * 8;

  for (int kb = 0; kb <= qb; ++kb) {
    const int kbase = kb * 64;
    const bool diag = (kb == qb);

    // ---- GEMM1: scores [16 x 64] ----
    v8f sc[4];
#pragma unroll
    for (int nt = 0; nt < 4; ++nt) {
      sc[nt] = v8f_zero();
#pragma unroll
      for (int ks = 0; ks < 4; ++ks) {
        const v16bf bk =
            load16(Kp + (size_t)(kbase + nt * 16 + lr) * HD + ks * 32 + lh * 16);
        sc[nt] = __builtin_amdgcn_wmma_f32_16x16x32_bf16(
            false, aq[ks], false, bk, (short)0, sc[nt], false, false);
      }
    }

    // ---- scale + causal mask + row max (rows = lh*8 + r) ----
    float rowmax[8];
#pragma unroll
    for (int r = 0; r < 8; ++r) rowmax[r] = -1e30f;
#pragma unroll
    for (int nt = 0; nt < 4; ++nt) {
      const int kc = kbase + nt * 16 + lr;
#pragma unroll
      for (int r = 0; r < 8; ++r) {
        float v = sc[nt][r] * scale;
        if (diag && kc > qrow0 + r) v = -1e30f;
        sc[nt][r] = v;
        rowmax[r] = fmaxf(rowmax[r], v);
      }
    }
#pragma unroll
    for (int r = 0; r < 8; ++r)
#pragma unroll
      for (int off = 8; off > 0; off >>= 1)
        rowmax[r] = fmaxf(rowmax[r], __shfl_xor(rowmax[r], off, 32));

    float corr[8], rowsum[8];
#pragma unroll
    for (int r = 0; r < 8; ++r) {
      const float mnew = fmaxf(m_r[r], rowmax[r]);
      corr[r] = __expf(m_r[r] - mnew);
      m_r[r] = mnew;
      rowsum[r] = 0.0f;
    }

    // ---- P = exp(sc - m): stage bf16 to per-wave LDS ----
#pragma unroll
    for (int nt = 0; nt < 4; ++nt)
#pragma unroll
      for (int r = 0; r < 8; ++r) {
        const float p = __expf(sc[nt][r] - m_r[r]);
        rowsum[r] += p;
        Plds[wave][lh * 8 + r][nt * 16 + lr] = (__bf16)p;
      }

#pragma unroll
    for (int r = 0; r < 8; ++r) {
#pragma unroll
      for (int off = 8; off > 0; off >>= 1)
        rowsum[r] += __shfl_xor(rowsum[r], off, 32);
      l_r[r] = l_r[r] * corr[r] + rowsum[r];
    }

    // rescale running O accumulators
#pragma unroll
    for (int nt8 = 0; nt8 < 8; ++nt8)
#pragma unroll
      for (int r = 0; r < 8; ++r) o[nt8][r] *= corr[r];

    // drain our LDS stores before re-reading P as A-fragments (same wave)
    asm volatile("s_wait_dscnt 0" ::: "memory");

    // ---- GEMM2: O[16 x 128] += P[16 x 64] @ V[64 x 128] ----
    v16bf ap[2];
#pragma unroll
    for (int ks2 = 0; ks2 < 2; ++ks2)
      ap[ks2] = load16(&Plds[wave][lr][ks2 * 32 + lh * 16]);
#pragma unroll
    for (int nt8 = 0; nt8 < 8; ++nt8)
#pragma unroll
      for (int ks2 = 0; ks2 < 2; ++ks2) {
        const v16bf bv = load16(
            Vp + (size_t)(nt8 * 16 + lr) * SEQLEN + kbase + ks2 * 32 + lh * 16);
        o[nt8] = __builtin_amdgcn_wmma_f32_16x16x32_bf16(
            false, ap[ks2], false, bv, (short)0, o[nt8], false, false);
      }
  }

  // ---- normalize and write bf16 [B*S, H] for the O-projection GEMM ----
  float rinv[8];
#pragma unroll
  for (int r = 0; r < 8; ++r) rinv[r] = 1.0f / l_r[r];
#pragma unroll
  for (int nt8 = 0; nt8 < 8; ++nt8)
#pragma unroll
    for (int r = 0; r < 8; ++r) {
      const size_t row = (size_t)b * SEQLEN + qrow0 + r;
      attn_out[row * HID + h * HD + nt8 * 16 + lr] = (__bf16)(o[nt8][r] * rinv[r]);
    }
}

// ---------------------------------------------------------------------------
// Host launcher
// ---------------------------------------------------------------------------
extern "C" void kernel_launch(void* const* d_in, const int* in_sizes, int n_in,
                              void* d_out, int out_size, void* d_ws, size_t ws_size,
                              hipStream_t stream) {
  (void)in_sizes; (void)n_in; (void)out_size; (void)ws_size;
  const float* hidden = (const float*)d_in[0];
  const float* w_pack = (const float*)d_in[1];
  const float* w_o    = (const float*)d_in[2];
  const int*   pos    = (const int*)d_in[3];
  float* out = (float*)d_out;

  char* ws = (char*)d_ws;
  size_t off = 0;
  auto take = [&](size_t bytes) -> char* {
    char* p = ws + off;
    off += (bytes + 255) & ~(size_t)255;
    return p;
  };
  __bf16* Xbf   = (__bf16*)take((size_t)MTOT * HID * 2);            //  32 MB
  __bf16* Wpbf  = (__bf16*)take((size_t)N_QKV * HID * 2);           //  96 MB
  __bf16* Wobf  = (__bf16*)take((size_t)HID * HID * 2);             //  32 MB
  __bf16* QKVbf = (__bf16*)take((size_t)MTOT * N_QKV * 2);          //  96 MB
  __bf16* Qbf   = (__bf16*)take((size_t)BATCH * NH * SEQLEN * HD * 2); // 32 MB
  __bf16* Kbf   = (__bf16*)take((size_t)BATCH * NH * SEQLEN * HD * 2); // 32 MB
  __bf16* Vtbf  = (__bf16*)take((size_t)BATCH * NH * HD * SEQLEN * 2); // 32 MB
  __bf16* Abf   = (__bf16*)take((size_t)MTOT * HID * 2);            //  32 MB

  cvt_f32_to_bf16<<<2048, 256, 0, stream>>>(hidden, Xbf, (size_t)MTOT * HID);
  cvt_f32_to_bf16<<<2048, 256, 0, stream>>>(w_pack, Wpbf, (size_t)N_QKV * HID);
  cvt_f32_to_bf16<<<2048, 256, 0, stream>>>(w_o, Wobf, (size_t)HID * HID);

  gemm_bf16_nt<__bf16><<<dim3(N_QKV / 128, MTOT / 128), 256, 0, stream>>>(
      Xbf, Wpbf, QKVbf, MTOT, N_QKV, HID);

  rope_split_kernel<<<BATCH * NH * SEQLEN, 128, 0, stream>>>(
      QKVbf, pos, Qbf, Kbf, Vtbf);

  flash_attn_kernel<<<dim3(SEQLEN / 64, NH, BATCH), 128, 0, stream>>>(
      Qbf, Kbf, Vtbf, Abf);

  gemm_bf16_nt<float><<<dim3(HID / 128, MTOT / 128), 256, 0, stream>>>(
      Abf, Wobf, out, MTOT, HID, HID);
}
